// VectorQuantizer_87531433493250
// MI455X (gfx1250) — compile-verified, matrix operand order checked by reading, not run on hardware
//
#include <hip/hip_runtime.h>

// ---------------------------------------------------------------------------
// VQ nearest-codebook search on MI455X (gfx1250, wave32, WMMA).
// dist(n,k) = ||c_k||^2 - 2 * dot(z_n, c_k)   (z-norm constant per row)
// dot via bf16 split precision: x = hi + lo;  hi*hi + hi*lo + lo*hi.
// Codebook tiles double-buffered in LDS via GLOBAL_LOAD_ASYNC_TO_LDS_B128.
// ---------------------------------------------------------------------------

typedef __attribute__((ext_vector_type(16))) __bf16 v16bf;
typedef __attribute__((ext_vector_type(8)))  float  v8f;
typedef __attribute__((ext_vector_type(4)))  float  v4f;

#define D_DIM   256
#define K_CODES 8192
#define N_ROWS  32768            // B*T = 8*4096
#define DCHUNKS 8                // 256 / 32
#define MTILES  (N_ROWS / 16)    // 2048
#define NT2_IT  (K_CODES / 32)   // 256 iterations, 2 code-tiles each
#define TILE_ELEMS 16384         // 32 frags x 512 bf16 (32 KB) per K-chunk

// ISA 16-bit fragment K-mapping: element e, lane-half h -> K index
__device__ __forceinline__ int kmap(int e, int h) {
    return ((e >> 3) << 4) + (h << 3) + (e & 7);
}

// ---------------------------------------------------------------------------
// Pack a row-major fp32 matrix [ntiles*16, 256] into WMMA-fragment-ordered
// bf16 hi/lo arrays. interleave=0: separate hi/lo arrays (A path).
// interleave=1: hi/lo fragments interleaved in one array so each 32-code
// K-chunk tile is a contiguous 32 KB block (B path, async-copy friendly).
// ---------------------------------------------------------------------------
__global__ void vq_pack_frags(const float* __restrict__ src,
                              __bf16* __restrict__ hi,
                              __bf16* __restrict__ lo,
                              int interleave) {
    int t    = blockIdx.x * blockDim.x + threadIdx.x;
    int lane = t & 31;
    int frag = t >> 5;               // frag = mt*8 + dc
    int dc   = frag & 7;
    int mt   = frag >> 3;
    int row  = mt * 16 + (lane & 15);
    int h    = lane >> 4;

    const float* p = src + row * D_DIM + dc * 32;
    v16bf vh, vl;
#pragma unroll
    for (int e = 0; e < 16; ++e) {
        float  x  = p[kmap(e, h)];
        __bf16 xh = (__bf16)x;
        float  r  = x - (float)xh;
        vh[e] = xh;
        vl[e] = (__bf16)r;
    }
    size_t hi_off = interleave ? ((size_t)frag * 2 + 0) * 512 : (size_t)frag * 512;
    size_t lo_off = interleave ? ((size_t)frag * 2 + 1) * 512 : (size_t)frag * 512;
    *(v16bf*)(hi + hi_off + lane * 16) = vh;
    *(v16bf*)(lo + lo_off + lane * 16) = vl;
}

// ---------------------------------------------------------------------------
// Codebook squared norms (fp32, one wave per code).
// ---------------------------------------------------------------------------
__global__ void vq_cb_norms(const float* __restrict__ cb,
                            float* __restrict__ out) {
    int t    = blockIdx.x * blockDim.x + threadIdx.x;
    int lane = t & 31;
    int code = t >> 5;
    const float* p = cb + (size_t)code * D_DIM + lane * 8;
    float s = 0.f;
#pragma unroll
    for (int i = 0; i < 8; ++i) s += p[i] * p[i];
#pragma unroll
    for (int m = 16; m >= 1; m >>= 1) s += __shfl_xor(s, m, 32);
    if (lane == 0) out[code] = s;
}

// Issue one 32 KB tile fill: 256 threads x 8 x b128, fully coalesced.
__device__ __forceinline__ void async_fill_tile(const __bf16* gbase,
                                                unsigned lds_base_b,
                                                int tid) {
    unsigned goff = (unsigned)tid * 16u;
    unsigned dst  = lds_base_b + (unsigned)tid * 16u;
#pragma unroll
    for (int i = 0; i < 8; ++i) {
        asm volatile("global_load_async_to_lds_b128 %0, %1, %2"
                     :
                     : "v"(dst + (unsigned)i * 4096u),
                       "v"(goff + (unsigned)i * 4096u),
                       "s"(gbase)
                     : "memory");
    }
}

// ---------------------------------------------------------------------------
// Main kernel: 8 waves/WG, each wave owns one 16-row tile; loop over all
// codes 32 at a time; codebook fragments double-buffered in LDS (2x32 KB)
// filled with async-to-LDS copies overlapped against WMMA compute.
// ---------------------------------------------------------------------------
__launch_bounds__(256, 1)
__global__ void vq_main(const __bf16* __restrict__ a_hi,
                        const __bf16* __restrict__ a_lo,
                        const __bf16* __restrict__ b_cat,
                        const float*  __restrict__ cnorm,
                        const float*  __restrict__ cb,
                        float* __restrict__ q_out,
                        float* __restrict__ idx_out) {
    __shared__ __bf16 lds_buf[2 * TILE_ELEMS];   // 64 KB double buffer

    const int tid  = threadIdx.x;
    const int lane = tid & 31;
    const int w    = tid >> 5;
    const int mt   = blockIdx.x * 8 + w;

    const unsigned lds0 = (unsigned)(uintptr_t)(void*)lds_buf; // raw LDS byte addr

    // A fragments (this wave's 16 rows) resident in registers.
    v16bf A_hi[DCHUNKS], A_lo[DCHUNKS];
#pragma unroll
    for (int dc = 0; dc < DCHUNKS; ++dc) {
        size_t off = ((size_t)mt * DCHUNKS + dc) * 512 + lane * 16;
        A_hi[dc] = *(const v16bf*)(a_hi + off);
        A_lo[dc] = *(const v16bf*)(a_lo + off);
    }

    float bestd[8];
    int   bestk[8];
#pragma unroll
    for (int r = 0; r < 8; ++r) { bestd[r] = 3.4e38f; bestk[r] = 0; }

    const int n = lane & 15;

    // Prologue: fill buffer 0 with tile 0.
    async_fill_tile(b_cat, lds0, tid);

    // unroll 1: keep one iteration's live ranges -> stay under 256 VGPRs and
    // avoid s_set_vgpr_msb churn in the hot loop.
#pragma unroll 1
    for (int nt2 = 0; nt2 < NT2_IT; ++nt2) {
        const int cur = nt2 & 1;

        // My async fills for the current buffer are done...
        asm volatile("s_wait_asynccnt 0" ::: "memory");
        // ...and after the barrier everyone's are, and everyone has finished
        // reading the other buffer (their dscnt waits precede their signal).
        __syncthreads();

        // Kick off next tile into the other buffer (overlaps compute below).
        if (nt2 + 1 < NT2_IT) {
            async_fill_tile(b_cat + (size_t)(nt2 + 1) * TILE_ELEMS,
                            lds0 + (unsigned)(cur ^ 1) * (TILE_ELEMS * 2u),
                            tid);
        }

        const __bf16* lbase = lds_buf + (size_t)cur * TILE_ELEMS;

        v8f acc0 = {}, acc1 = {};
#pragma unroll
        for (int dc = 0; dc < DCHUNKS; ++dc) {
            v16bf b0h = *(const v16bf*)(lbase + (dc * 2 + 0) * 512 + lane * 16);
            v16bf b0l = *(const v16bf*)(lbase + (dc * 2 + 1) * 512 + lane * 16);
            v16bf b1h = *(const v16bf*)(lbase + (16 + dc * 2 + 0) * 512 + lane * 16);
            v16bf b1l = *(const v16bf*)(lbase + (16 + dc * 2 + 1) * 512 + lane * 16);

            acc0 = __builtin_amdgcn_wmma_f32_16x16x32_bf16(
                false, A_hi[dc], false, b0h, (short)0, acc0, false, false);
            acc0 = __builtin_amdgcn_wmma_f32_16x16x32_bf16(
                false, A_hi[dc], false, b0l, (short)0, acc0, false, false);
            acc0 = __builtin_amdgcn_wmma_f32_16x16x32_bf16(
                false, A_lo[dc], false, b0h, (short)0, acc0, false, false);

            acc1 = __builtin_amdgcn_wmma_f32_16x16x32_bf16(
                false, A_hi[dc], false, b1h, (short)0, acc1, false, false);
            acc1 = __builtin_amdgcn_wmma_f32_16x16x32_bf16(
                false, A_hi[dc], false, b1l, (short)0, acc1, false, false);
            acc1 = __builtin_amdgcn_wmma_f32_16x16x32_bf16(
                false, A_lo[dc], false, b1h, (short)0, acc1, false, false);
        }

        const int code0 = nt2 * 32 + n;
        const int code1 = code0 + 16;
        const float cn0 = cnorm[code0];
        const float cn1 = cnorm[code1];
#pragma unroll
        for (int r = 0; r < 8; ++r) {
            float d0 = cn0 - 2.0f * acc0[r];
            if (d0 < bestd[r]) { bestd[r] = d0; bestk[r] = code0; }
            float d1 = cn1 - 2.0f * acc1[r];
            if (d1 < bestd[r]) { bestd[r] = d1; bestk[r] = code1; }
        }
    }

    // Cross-lane argmin over the 16 lanes that hold each C-matrix row.
    int bkFin[8];
#pragma unroll
    for (int r = 0; r < 8; ++r) {
        float bd = bestd[r];
        int   bk = bestk[r];
#pragma unroll
        for (int m = 1; m < 16; m <<= 1) {
            float od = __shfl_xor(bd, m, 32);
            int   ok = __shfl_xor(bk, m, 32);
            if (od < bd || (od == bd && ok < bk)) { bd = od; bk = ok; }
        }
        bkFin[r] = bk;
        int row = mt * 16 + (lane >> 4) * 8 + r;
        if ((lane & 15) == 0) idx_out[row] = (float)bk;
    }

    // Gather winning codebook rows (fp32 originals) into q_out.
#pragma unroll
    for (int row = 0; row < 16; ++row) {
        int bk = __shfl(bkFin[row & 7], (row < 8) ? 0 : 16, 32);
        const float* srcp = cb    + (size_t)bk * D_DIM + lane * 8;
        float*       dstp = q_out + ((size_t)mt * 16 + row) * D_DIM + lane * 8;
        *(v4f*)(dstp)     = *(const v4f*)(srcp);
        *(v4f*)(dstp + 4) = *(const v4f*)(srcp + 4);
    }
}

// ---------------------------------------------------------------------------
extern "C" void kernel_launch(void* const* d_in, const int* in_sizes, int n_in,
                              void* d_out, int out_size, void* d_ws, size_t ws_size,
                              hipStream_t stream) {
    const float* z  = (const float*)d_in[0];   // [N_ROWS, 256]
    const float* cb = (const float*)d_in[1];   // [K_CODES, 256]

    char* ws = (char*)d_ws;
    size_t sz_a = (size_t)N_ROWS  * D_DIM * sizeof(__bf16);      // 16 MB each
    size_t sz_b = (size_t)K_CODES * D_DIM * 2 * sizeof(__bf16);  // 8 MB hi+lo
    __bf16* a_hi  = (__bf16*)(ws);
    __bf16* a_lo  = (__bf16*)(ws + sz_a);
    __bf16* b_cat = (__bf16*)(ws + 2 * sz_a);
    float*  cnm   = (float*)(ws + 2 * sz_a + sz_b);

    float* q_out   = (float*)d_out;                           // N*D fp32
    float* idx_out = (float*)d_out + (size_t)N_ROWS * D_DIM;  // N indices

    // Pack z (separate hi/lo): (N/16)*8 frags * 32 lanes threads
    vq_pack_frags<<<(MTILES * DCHUNKS * 32) / 256, 256, 0, stream>>>(z, a_hi, a_lo, 0);
    // Pack codebook (interleaved hi/lo -> contiguous 32 KB K-chunk tiles)
    vq_pack_frags<<<((K_CODES / 16) * DCHUNKS * 32) / 256, 256, 0, stream>>>(cb, b_cat, b_cat, 1);
    // Codebook norms: one wave per code
    vq_cb_norms<<<(K_CODES * 32) / 256, 256, 0, stream>>>(cb, cnm);
    // Main: 256 WGs x 8 waves, each wave one 16-row tile
    vq_main<<<MTILES / 8, 256, 0, stream>>>(a_hi, a_lo, b_cat, cnm, cb,
                                            q_out, idx_out);
}
